// ParallelCA_79757542687139
// MI455X (gfx1250) — matrix operand order repaired and checked by reading, not compiled, not run on hardware
//
#include <hip/hip_runtime.h>
#include <hip/hip_bf16.h>
#include <math.h>

typedef __attribute__((ext_vector_type(16))) _Float16 v16h;
typedef __attribute__((ext_vector_type(8)))  float    v8f;

static constexpr int NB  = 8;     // batch
static constexpr int CCH = 256;   // channels
static constexpr int HW  = 1024;  // H*W
static constexpr int CL  = 16;    // projected channels
static constexpr int CLP = 32;    // padded (zeros) to one WMMA K

union V16H { v16h h; uint4 u[2]; _Float16 e[16]; };

// ---------------- spatial attention: per-pixel channel dot ----------------
__global__ __launch_bounds__(256)
void k_sa_dot(const float* __restrict__ fp1, const float* __restrict__ fp2,
              const float* __restrict__ cw1, const float* __restrict__ cb1,
              const float* __restrict__ cw2, const float* __restrict__ cb2,
              float* __restrict__ c1, float* __restrict__ c2) {
  int p = blockIdx.x * 256 + threadIdx.x;
  int n = blockIdx.y;
  const float* b1 = fp1 + (size_t)n * CCH * HW + p;
  const float* b2 = fp2 + (size_t)n * CCH * HW + p;
  float a1 = 0.f, a2 = 0.f;
  for (int c = 0; c < CCH; ++c) {
    a1 += cw1[c] * b1[(size_t)c * HW];
    a2 += cw2[c] * b2[(size_t)c * HW];
  }
  c1[n * HW + p] = a1 + cb1[0];
  c2[n * HW + p] = a2 + cb2[0];
}

// ---------------- softmax over HW (in place -> weights) ----------------
__global__ __launch_bounds__(1024)
void k_softmax(float* __restrict__ c1, float* __restrict__ c2) {
  __shared__ float sm[1024];
  int n = blockIdx.x, t = threadIdx.x;
  float v1 = c1[n * HW + t], v2 = c2[n * HW + t];

  sm[t] = v1; __syncthreads();
  for (int s = 512; s > 0; s >>= 1) { if (t < s) sm[t] = fmaxf(sm[t], sm[t + s]); __syncthreads(); }
  float m1 = sm[0]; __syncthreads();
  sm[t] = v2; __syncthreads();
  for (int s = 512; s > 0; s >>= 1) { if (t < s) sm[t] = fmaxf(sm[t], sm[t + s]); __syncthreads(); }
  float m2 = sm[0]; __syncthreads();

  float e1 = expf(v1 - m1), e2 = expf(v2 - m2);
  sm[t] = e1; __syncthreads();
  for (int s = 512; s > 0; s >>= 1) { if (t < s) sm[t] += sm[t + s]; __syncthreads(); }
  float s1 = sm[0]; __syncthreads();
  sm[t] = e2; __syncthreads();
  for (int s = 512; s > 0; s >>= 1) { if (t < s) sm[t] += sm[t + s]; __syncthreads(); }
  float s2 = sm[0];

  c1[n * HW + t] = e1 / s1;
  c2[n * HW + t] = e2 / s2;
}

// ---------------- s = w * fp (f16 copy; fp32 to d_out on last iter) ----------------
__global__ __launch_bounds__(256)
void k_scale(const float* __restrict__ fp, const float* __restrict__ w,
             _Float16* __restrict__ sh, float* __restrict__ sout) {
  size_t idx = (size_t)blockIdx.x * 256 + threadIdx.x;   // over NB*CCH*HW
  int n = (int)(idx >> 18);                              // /(CCH*HW)
  int p = (int)(idx & (HW - 1));
  float s = w[n * HW + p] * fp[idx];
  sh[idx] = (_Float16)s;
  if (sout) sout[idx] = s;
}

// -------- f_clT[n,p,l] = pb[l] + sum_c pw[l,c]*s[n,c,p];  l=16..31 zero-padded -----
// Transposed [p][l] layout => WMMA fragments become contiguous-K b128 loads.
__global__ __launch_bounds__(256)
void k_proj(const _Float16* __restrict__ sh, const float* __restrict__ pw,
            const float* __restrict__ pb, _Float16* __restrict__ fclT) {
  int p = blockIdx.x * 256 + threadIdx.x;
  int l = blockIdx.y;            // 0..31 (>=16 => zero pad)
  int n = blockIdx.z;
  float acc = 0.f;
  if (l < CL) {
    const _Float16* s = sh + (size_t)n * CCH * HW + p;
    const float* wr = pw + l * CCH;
    for (int c = 0; c < CCH; ++c) acc += wr[c] * (float)s[(size_t)c * HW];
    acc += pb[l];
  }
  fclT[((size_t)n * HW + p) * CLP + l] = (_Float16)acc;
}

// ------- A[n,i,j] = tanh(sum_l f2cl[l,i]*f1cl[l,j]); store A and A^T (f16) --------
// f_clT layout: [n][p][CLP]; A-op(m,k)=f2clT[i0+m][k], B-op(k,n')=f1clT[j0+n'][k].
__global__ __launch_bounds__(256)
void k_Agemm(const _Float16* __restrict__ f1T, const _Float16* __restrict__ f2T,
             _Float16* __restrict__ Ah, _Float16* __restrict__ At) {
  int lane = threadIdx.x & 31;
  int wv   = threadIdx.x >> 5;
  int jt = blockIdx.x * 8 + wv;  // j-tile 0..63
  int it = blockIdx.y;           // i-tile 0..63
  int n  = blockIdx.z;
  int i0 = it * 16, j0 = jt * 16;
  int m    = lane & 15;
  int kb   = (lane >> 4) << 3;   // A-frag K base: 0 or 8
  int koff = (lane >> 4) << 4;   // unused (B-frag also uses kb)

  const _Float16* arow = f2T + ((size_t)n * HW + i0 + m) * CLP;
  const _Float16* brow = f1T + ((size_t)n * HW + j0 + m) * CLP;

  V16H a, b;
  a.u[0] = *(const uint4*)(arow + kb);        // K = kb .. kb+7
  a.u[1] = *(const uint4*)(arow + 16 + kb);   // K = 16+kb .. 16+kb+7 (zeros)
  b.u[0] = *(const uint4*)(brow + kb);        // K = kb .. kb+7
  b.u[1] = *(const uint4*)(brow + 16 + kb);   // K = 16+kb .. 16+kb+7 (zeros)

  v8f cz = {};
  v8f d = __builtin_amdgcn_wmma_f32_16x16x32_f16(false, a.h, false, b.h,
                                                 (short)0, cz, false, false);
#pragma unroll
  for (int v = 0; v < 8; ++v) {
    int row  = i0 + v + ((lane >> 4) << 3);
    int colj = j0 + m;
    _Float16 th = (_Float16)tanhf(d[v]);
    Ah[((size_t)n << 20) + (size_t)row * HW + colj] = th;
    At[((size_t)n << 20) + (size_t)colj * HW + row] = th;
  }
}

// ----- hat[c,i] = sum_j s[c,j] * Asel[i,j]; 2x2 register tile (32x32 per wave) -----
__global__ __launch_bounds__(256)
void k_hat(const _Float16* __restrict__ sh, const _Float16* __restrict__ Asel,
           float* __restrict__ hat) {
  int lane = threadIdx.x & 31;
  int wv   = threadIdx.x >> 5;
  int ntp = blockIdx.x * 8 + wv; // i-tile pair 0..31
  int mtp = blockIdx.y;          // c-tile pair 0..7
  int n   = blockIdx.z;
  int c0 = mtp * 32, i0 = ntp * 32;
  int m    = lane & 15;
  int kb   = (lane >> 4) << 3;
  int koff = (lane >> 4) << 4;

  const _Float16* srow0 = sh   + ((size_t)n * CCH + c0 + m) * HW;
  const _Float16* srow1 = srow0 + (size_t)16 * HW;
  const _Float16* arow0 = Asel + ((size_t)n << 20) + (size_t)(i0 + m) * HW;
  const _Float16* arow1 = arow0 + (size_t)16 * HW;

  v8f acc00 = {}, acc01 = {}, acc10 = {}, acc11 = {};
#pragma unroll 2
  for (int kt = 0; kt < 64; ++kt) {
    int j0 = kt * 32;
    V16H a0, a1, b0, b1;
    a0.u[0] = *(const uint4*)(srow0 + j0 + kb);
    a0.u[1] = *(const uint4*)(srow0 + j0 + 16 + kb);
    a1.u[0] = *(const uint4*)(srow1 + j0 + kb);
    a1.u[1] = *(const uint4*)(srow1 + j0 + 16 + kb);
    b0.u[0] = *(const uint4*)(arow0 + j0 + kb);
    b0.u[1] = *(const uint4*)(arow0 + j0 + 16 + kb);
    b1.u[0] = *(const uint4*)(arow1 + j0 + kb);
    b1.u[1] = *(const uint4*)(arow1 + j0 + 16 + kb);
    acc00 = __builtin_amdgcn_wmma_f32_16x16x32_f16(false, a0.h, false, b0.h, (short)0, acc00, false, false);
    acc01 = __builtin_amdgcn_wmma_f32_16x16x32_f16(false, a0.h, false, b1.h, (short)0, acc01, false, false);
    acc10 = __builtin_amdgcn_wmma_f32_16x16x32_f16(false, a1.h, false, b0.h, (short)0, acc10, false, false);
    acc11 = __builtin_amdgcn_wmma_f32_16x16x32_f16(false, a1.h, false, b1.h, (short)0, acc11, false, false);
  }
#pragma unroll
  for (int v = 0; v < 8; ++v) {
    int r0 = c0 + v + ((lane >> 4) << 3);
    int r1 = r0 + 16;
    size_t base0 = ((size_t)n * CCH + r0) * HW;
    size_t base1 = ((size_t)n * CCH + r1) * HW;
    hat[base0 + i0 + m]      = acc00[v];
    hat[base0 + i0 + m + 16] = acc01[v];
    hat[base1 + i0 + m]      = acc10[v];
    hat[base1 + i0 + m + 16] = acc11[v];
  }
}

// ---------------- channel L2-norm; residual ReLU or final output -------------------
__global__ __launch_bounds__(256)
void k_norm(const float* __restrict__ hat, float* __restrict__ fp,
            float* __restrict__ outh, int last) {
  int i = blockIdx.x * 256 + threadIdx.x;
  int n = blockIdx.y;
  const float* base = hat + (size_t)n * CCH * HW + i;
  float ss = 0.f;
  for (int c = 0; c < CCH; ++c) { float v = base[(size_t)c * HW]; ss += v * v; }
  float inv = 1.0f / fmaxf(sqrtf(ss), 1e-12f);
  if (last) {
    float* ob = outh + (size_t)n * CCH * HW + i;
    for (int c = 0; c < CCH; ++c) ob[(size_t)c * HW] = base[(size_t)c * HW] * inv;
  } else {
    float* fb = fp + (size_t)n * CCH * HW + i;
    for (int c = 0; c < CCH; ++c) {
      float h = base[(size_t)c * HW] * inv;
      fb[(size_t)c * HW] = fmaxf(h + fb[(size_t)c * HW], 0.f);
    }
  }
}

extern "C" void kernel_launch(void* const* d_in, const int* in_sizes, int n_in,
                              void* d_out, int out_size, void* d_ws, size_t ws_size,
                              hipStream_t stream) {
  const float* f1  = (const float*)d_in[0];
  const float* f2  = (const float*)d_in[1];
  const float* pw1 = (const float*)d_in[2];
  const float* pb1 = (const float*)d_in[3];
  const float* pw2 = (const float*)d_in[4];
  const float* pb2 = (const float*)d_in[5];
  const float* cw1 = (const float*)d_in[6];
  const float* cb1 = (const float*)d_in[7];
  const float* cw2 = (const float*)d_in[8];
  const float* cb2 = (const float*)d_in[9];
  float* out = (float*)d_out;

  const size_t SZ = (size_t)NB * CCH * HW;  // 2,097,152 elements per tensor

  // Workspace carve-up (all 256B aligned by construction)
  char* w = (char*)d_ws;
  size_t off = 0;
  auto carve = [&](size_t bytes) { char* p = w + off; off += (bytes + 255) & ~(size_t)255; return p; };
  float*    fp1  = (float*)   carve(SZ * 4);
  float*    fp2  = (float*)   carve(SZ * 4);
  float*    c1   = (float*)   carve((size_t)NB * HW * 4);
  float*    c2   = (float*)   carve((size_t)NB * HW * 4);
  _Float16* s1h  = (_Float16*)carve(SZ * 2);
  _Float16* s2h  = (_Float16*)carve(SZ * 2);
  _Float16* f1T  = (_Float16*)carve((size_t)NB * HW * CLP * 2);
  _Float16* f2T  = (_Float16*)carve((size_t)NB * HW * CLP * 2);
  _Float16* Ah   = (_Float16*)carve((size_t)NB * HW * HW * 2);
  _Float16* At   = (_Float16*)carve((size_t)NB * HW * HW * 2);
  float*    hat1 = (float*)   carve(SZ * 4);
  float*    hat2 = (float*)   carve(SZ * 4);
  (void)ws_size; (void)in_sizes; (void)n_in; (void)out_size;

  hipMemcpyAsync(fp1, f1, SZ * 4, hipMemcpyDeviceToDevice, stream);
  hipMemcpyAsync(fp2, f2, SZ * 4, hipMemcpyDeviceToDevice, stream);

  for (int it = 0; it < 5; ++it) {
    const int last = (it == 4);

    k_sa_dot<<<dim3(HW / 256, NB), 256, 0, stream>>>(fp1, fp2, cw1, cb1, cw2, cb2, c1, c2);
    k_softmax<<<NB, 1024, 0, stream>>>(c1, c2);

    k_scale<<<(unsigned)(SZ / 256), 256, 0, stream>>>(fp1, c1, s1h, last ? (out + 2 * SZ) : nullptr);
    k_scale<<<(unsigned)(SZ / 256), 256, 0, stream>>>(fp2, c2, s2h, last ? (out + 3 * SZ) : nullptr);

    k_proj<<<dim3(HW / 256, CLP, NB), 256, 0, stream>>>(s1h, pw1, pb1, f1T);
    k_proj<<<dim3(HW / 256, CLP, NB), 256, 0, stream>>>(s2h, pw2, pb2, f2T);

    k_Agemm<<<dim3(8, 64, NB), 256, 0, stream>>>(f1T, f2T, Ah, At);

    // f1_hat[c,i] = sum_j s1[c,j]*A[i,j]   -> Asel = A   (row i, contiguous j)
    // f2_hat[c,i] = sum_j s2[c,j]*A[j,i]   -> Asel = A^T (row i, contiguous j)
    k_hat<<<dim3(4, 8, NB), 256, 0, stream>>>(s1h, Ah, hat1);
    k_hat<<<dim3(4, 8, NB), 256, 0, stream>>>(s2h, At, hat2);

    k_norm<<<dim3(HW / 256, NB), 256, 0, stream>>>(hat1, fp1, out + 0 * SZ, last);
    k_norm<<<dim3(HW / 256, NB), 256, 0, stream>>>(hat2, fp2, out + 1 * SZ, last);
  }
}
